// UniPhyBlock_89335319757464
// MI455X (gfx1250) — matrix operand order repaired and checked by reading, not compiled, not run on hardware
//
#include <hip/hip_runtime.h>
#include <cstdint>
#include <cstddef>

// ---------------------------------------------------------------------------
// Types / WMMA helpers (gfx1250 wave32, v_wmma_f32_16x16x32_f16)
// ---------------------------------------------------------------------------
typedef _Float16 half_t;
typedef __attribute__((ext_vector_type(16))) _Float16 v16h;
typedef __attribute__((ext_vector_type(8)))  float v8f;
typedef __attribute__((ext_vector_type(4)))  float f32x4;

#define B_   2
#define T_   16
#define D_   64
#define HW_  4096
#define C2_  128
#define NT_  131072   /* B*T*HW tokens */
#define HID_ 256
#define NE_  4
#define BTD_ 2048     /* B*T*D */
#define OUTPART_ 8388608  /* B*T*D*HW */
#define CPX_ 128      /* conv pixel tile */
#define BSTR_ 136     /* conv LDS B-tile stride (halves), pads banks */

union V16HU { v16h v; f32x4 q[2]; };

__device__ __forceinline__ v8f wmma16(v16h a, v16h b, v8f c) {
  return __builtin_amdgcn_wmma_f32_16x16x32_f16(false, a, false, b, (short)0, c,
                                                false, false);
}

// A fragment: 16x32 f16 tile, row-major with `stride` halves per row.
// ISA layout: lane<16 -> M=lane, K={0..7,16..23}; lane>=16 -> K={8..15,24..31}
__device__ __forceinline__ v16h load_a_frag(const half_t* tile, int stride,
                                            int ko, int lane) {
  int m  = lane & 15;
  int kb = (lane >> 4) * 8;
  const half_t* p = tile + m * stride + ko + kb;
  V16HU u;
  u.q[0] = *(const f32x4*)(p);
  u.q[1] = *(const f32x4*)(p + 16);
  return u.v;
}

// B fragment: 32x16 f16 tile from a column-major matrix BT[col*stride + k].
// ISA layout: lane<16 -> col=lane, K={0..7,16..23}; lane>=16 -> K={8..15,24..31}
__device__ __forceinline__ v16h load_b_frag(const half_t* bt, int stride,
                                            int col0, int ko, int lane) {
  int c  = col0 + (lane & 15);
  int kb = (lane >> 4) * 8;
  const half_t* p = bt + c * stride + ko + kb;
  V16HU u;
  u.q[0] = *(const f32x4*)(p);
  u.q[1] = *(const f32x4*)(p + 16);
  return u.v;
}

// Async global->LDS 16-byte copy (ASYNCcnt-tracked, no VGPR bounce)
__device__ __forceinline__ void async_copy_b128(void* lds_dst, const void* gsrc) {
  unsigned ldsoff = (unsigned)(uintptr_t)lds_dst;            // low 32 = LDS offset
  unsigned long long ga = (unsigned long long)(uintptr_t)gsrc;
  asm volatile("global_load_async_to_lds_b128 %0, %1, off"
               :: "v"(ldsoff), "v"(ga) : "memory");
}
__device__ __forceinline__ void async_wait0() {
  asm volatile("s_wait_asynccnt 0" ::: "memory");
}

__device__ __forceinline__ float softplusf(float x) {
  return (x > 20.f) ? x : log1pf(expf(x));
}
__device__ __forceinline__ float sigmoidf_(float x) {
  return 1.f / (1.f + expf(-x));
}
__device__ __forceinline__ float geluf_(float x) {
  float x3 = x * x * x;
  float t  = tanhf(0.7978845608028654f * (x + 0.044715f * x3));
  return 0.5f * x * (1.f + t);
}
__device__ __forceinline__ unsigned pcg_hash(unsigned v) {
  unsigned s = v * 747796405u + 2891336453u;
  unsigned w = ((s >> ((s >> 28) + 4u)) ^ s) * 277803737u;
  return (w >> 22) ^ w;
}

// ---------------------------------------------------------------------------
// 1) LayerNorm over channels (2D=128) at each (bt, hw).
//    Writes TOKEN-MAJOR f16 xnT (n, c) so conv B-tiles are contiguous rows.
// ---------------------------------------------------------------------------
__global__ __launch_bounds__(256) void k_ln1(const float* __restrict__ x_re,
                                             const float* __restrict__ x_im,
                                             const float* __restrict__ gamma,
                                             const float* __restrict__ beta,
                                             half_t* __restrict__ xnT) {
  int i  = blockIdx.x * 256 + threadIdx.x;     // over B*T*HW
  int bt = i >> 12;
  int hw = i & (HW_ - 1);
  float s = 0.f, ss = 0.f;
  #pragma unroll 8
  for (int c = 0; c < C2_; ++c) {
    float v = (c < D_) ? x_re[(bt * D_ + c) * HW_ + hw]
                       : x_im[(bt * D_ + (c - D_)) * HW_ + hw];
    s += v; ss += v * v;
  }
  float mean = s * (1.f / C2_);
  float var  = ss * (1.f / C2_) - mean * mean;
  float inv  = rsqrtf(var + 1e-5f);
  half_t row[C2_];
  #pragma unroll
  for (int c = 0; c < C2_; ++c) {
    float v = (c < D_) ? x_re[(bt * D_ + c) * HW_ + hw]
                       : x_im[(bt * D_ + (c - D_)) * HW_ + hw];
    row[c] = (half_t)((v - mean) * inv * gamma[c] + beta[c]);
  }
  f32x4* dst = (f32x4*)(xnT + (size_t)i * C2_);
  const f32x4* srcv = (const f32x4*)row;
  #pragma unroll
  for (int k = 0; k < 16; ++k) dst[k] = srcv[k];
}

// ---------------------------------------------------------------------------
// 2) Conv as 9 per-tap WMMA GEMMs (K=128 ci). Double-buffered per-tap LDS
//    B tiles filled by async-to-LDS; next tap's DMA overlaps current WMMAs.
//    grid = (px_tile=32, bt=32); wave w owns co-tile 16 rows.
// ---------------------------------------------------------------------------
#define CONV_SMEM (2 * CPX_ * BSTR_ * 2)   /* 69632 B dynamic LDS */
__global__ __launch_bounds__(256) void k_conv(const half_t* __restrict__ xnT,
                                              const half_t* __restrict__ WmT,
                                              const float* __restrict__ bias,
                                              const float* __restrict__ metric,
                                              const float* __restrict__ x_re,
                                              const float* __restrict__ x_im,
                                              float* __restrict__ x2) {
  extern __shared__ char csmem[];
  half_t* buf0 = (half_t*)csmem;
  half_t* buf1 = buf0 + CPX_ * BSTR_;
  int tid = threadIdx.x, lane = tid & 31, wave = tid >> 5;
  int px0 = blockIdx.x * CPX_;
  int bt  = blockIdx.y;
  int co0 = wave * 16;

  auto zero_buf = [&](half_t* b) {
    f32x4 z = {0.f, 0.f, 0.f, 0.f};
    f32x4* d = (f32x4*)b;
    for (int i = tid; i < CPX_ * BSTR_ / 8; i += 256) d[i] = z;
  };
  auto issue_tap = [&](half_t* b, int tap) {
    int kh = tap / 3 - 1, kw = tap % 3 - 1;
    for (int it = 0; it < 8; ++it) {       // 2048 x 16B chunks, border-masked
      int idx = it * 256 + tid;
      int px = idx >> 4, seg = idx & 15;
      int h = (px0 + px) >> 6, w = (px0 + px) & 63;
      int hh = h + kh, ww = w + kw;
      if (hh >= 0 && hh < 64 && ww >= 0 && ww < 64) {
        const half_t* gp = xnT + (size_t)(bt * HW_ + hh * 64 + ww) * C2_ + seg * 8;
        async_copy_b128(b + px * BSTR_ + seg * 8, gp);
      }
    }
  };

  v8f acc[8] = {};
  // prologue: stage tap 0
  zero_buf(buf0);
  __syncthreads();                          // zeros visible before DMA lands
  issue_tap(buf0, 0);
  async_wait0();
  __syncthreads();

  for (int tap = 0; tap < 9; ++tap) {
    half_t* cur  = (tap & 1) ? buf1 : buf0;
    half_t* next = (tap & 1) ? buf0 : buf1;
    if (tap + 1 < 9) {
      zero_buf(next);                       // next buffer's readers done (barrier)
      __syncthreads();                      // zeros before async writes
      issue_tap(next, tap + 1);             // DMA overlaps compute below
    }
    const half_t* wtap = WmT + (size_t)tap * C2_ * C2_ + (size_t)co0 * C2_;
    for (int ko = 0; ko < C2_; ko += 32) {
      v16h af = load_a_frag(wtap, C2_, ko, lane);
      #pragma unroll
      for (int j = 0; j < 8; ++j) {
        v16h bf = load_b_frag(cur, BSTR_, j * 16, ko, lane);
        acc[j] = wmma16(af, bf, acc[j]);
      }
    }
    async_wait0();                          // next tile landed
    __syncthreads();
  }
  // epilogue: xs*softplus(metric)+bias + residual input
  #pragma unroll
  for (int j = 0; j < 8; ++j)
    #pragma unroll
    for (int r = 0; r < 8; ++r) {
      int co = co0 + r + (lane >> 4) * 8;
      int px = px0 + j * 16 + (lane & 15);
      float sp  = softplusf(metric[px]);
      float xin = (co < D_) ? x_re[(bt * D_ + co) * HW_ + px]
                            : x_im[(bt * D_ + (co - D_)) * HW_ + px];
      x2[(bt * C2_ + co) * HW_ + px] = xin + acc[j][r] * sp + bias[co];
    }
}

// Weight prep for conv: WmT[tap][co][ci] = conv_w[co][ci][kh][kw]
__global__ void k_trans_convw(const float* __restrict__ w, half_t* __restrict__ WmT) {
  int i = blockIdx.x * 256 + threadIdx.x;      // 9*128*128
  if (i >= 9 * C2_ * C2_) return;
  int tap = i >> 14;
  int co = (i >> 7) & 127;
  int ci = i & 127;
  WmT[i] = (half_t)w[(co * C2_ + ci) * 9 + tap];
}

// ---------------------------------------------------------------------------
// 3) Pack x2 (bt,c,hw) f32 -> token-major f16 feat (N x 128), LDS transpose
// ---------------------------------------------------------------------------
__global__ __launch_bounds__(256) void k_pack(const float* __restrict__ x2,
                                              half_t* __restrict__ feat) {
  __shared__ float lds[64 * 130];
  int hw0 = blockIdx.x * 64;
  int bt  = blockIdx.y;
  for (int it = 0; it < 32; ++it) {
    int idx = threadIdx.x + it * 256;
    int c = idx >> 6, hwl = idx & 63;
    lds[hwl * 130 + c] = x2[(bt * C2_ + c) * HW_ + hw0 + hwl];
  }
  __syncthreads();
  for (int it = 0; it < 32; ++it) {
    int idx = threadIdx.x + it * 256;
    int hwl = idx >> 7, c = idx & 127;
    feat[((size_t)(bt * HW_ + hw0 + hwl)) * C2_ + c] = (half_t)lds[hwl * 130 + c];
  }
}

// ---------------------------------------------------------------------------
// 4) Generic WMMA GEMM: C(N x 128) = A(N x 128, f16) * M(128x128)
//    BT column-major f16 staged in LDS via async-to-LDS.
// ---------------------------------------------------------------------------
__global__ __launch_bounds__(256) void k_gemm128(const half_t* __restrict__ A,
                                                 const half_t* __restrict__ BT,
                                                 float* __restrict__ C) {
  __shared__ half_t bts[C2_ * C2_];
  for (int i = threadIdx.x; i < 2048; i += 256)
    async_copy_b128(bts + i * 8, BT + (size_t)i * 8);
  int lane = threadIdx.x & 31;
  int wave = threadIdx.x >> 5;
  size_t row0 = (size_t)blockIdx.x * 128 + wave * 16;
  const half_t* a = A + row0 * C2_;
  __builtin_prefetch(a, 0, 3);
  __builtin_prefetch(a + 8 * C2_, 0, 3);
  async_wait0();
  __syncthreads();
  v8f acc[8] = {};
  for (int ko = 0; ko < 128; ko += 32) {
    v16h af = load_a_frag(a, C2_, ko, lane);
    #pragma unroll
    for (int j = 0; j < 8; ++j) {
      v16h bf = load_b_frag(bts, C2_, j * 16, ko, lane);
      acc[j] = wmma16(af, bf, acc[j]);
    }
  }
  #pragma unroll
  for (int j = 0; j < 8; ++j)
    #pragma unroll
    for (int r = 0; r < 8; ++r) {
      size_t row = row0 + r + (lane >> 4) * 8;
      int col = j * 16 + (lane & 15);
      C[row * C2_ + col] = acc[j][r];
    }
}

// ---------------------------------------------------------------------------
// 5) Build 128x128 real block matrix for complex GEMM (col-major f16)
// ---------------------------------------------------------------------------
__global__ void k_build_cmat(const float* __restrict__ Mre,
                             const float* __restrict__ Mim,
                             half_t* __restrict__ outT) {
  int i = blockIdx.x * 256 + threadIdx.x;
  if (i >= C2_ * C2_) return;
  int col = i >> 7, k = i & 127;
  int d = k & 63, c = col & 63;
  float v;
  if (k < 64 && col < 64)        v =  Mre[d * D_ + c];
  else if (k >= 64 && col < 64)  v = -Mim[d * D_ + c];
  else if (k < 64)               v =  Mim[d * D_ + c];
  else                           v =  Mre[d * D_ + c];
  outT[col * C2_ + k] = (half_t)v;
}

// Transpose-convert expert weights: W (E,R,C) f32 -> WT (E,C,R) f16
__global__ void k_trans_w(const float* __restrict__ W, half_t* __restrict__ WT,
                          int R, int Ccols, int total) {
  int i = blockIdx.x * 256 + threadIdx.x;
  if (i >= total) return;
  int e = i / (R * Ccols);
  int r = (i / Ccols) % R;
  int c = i % Ccols;
  WT[((size_t)e * Ccols + c) * R + r] = (half_t)W[i];
}

// ---------------------------------------------------------------------------
// 6) Spatial mean of x_eigen -> x_mean (B,T,D) complex
// ---------------------------------------------------------------------------
__global__ __launch_bounds__(256) void k_mean(const float* __restrict__ xeig,
                                              float* __restrict__ xm) {
  int bt = blockIdx.x;
  int d = threadIdx.x & 63, g = threadIdx.x >> 6;
  float sr = 0.f, si = 0.f;
  for (int hw = g; hw < HW_; hw += 4) {
    size_t n = (size_t)bt * HW_ + hw;
    sr += xeig[n * C2_ + d];
    si += xeig[n * C2_ + 64 + d];
  }
  __shared__ float red[2][256];
  red[0][threadIdx.x] = sr; red[1][threadIdx.x] = si;
  __syncthreads();
  if (g == 0) {
    for (int gg = 1; gg < 4; ++gg) { sr += red[0][gg * 64 + d]; si += red[1][gg * 64 + d]; }
    xm[(bt * D_ + d) * 2 + 0] = sr * (1.f / HW_);
    xm[(bt * D_ + d) * 2 + 1] = si * (1.f / HW_);
  }
}

// 7) X_flux = x_mean @ (Win_re + i Win_im)
__global__ void k_xflux(const float* __restrict__ xm,
                        const float* __restrict__ Wre,
                        const float* __restrict__ Wim,
                        float* __restrict__ Xf) {
  int i = blockIdx.x * 256 + threadIdx.x;
  if (i >= BTD_) return;
  int e = i & 63, bt = i >> 6;
  float ar = 0.f, ai = 0.f;
  for (int d = 0; d < D_; ++d) {
    float xr = xm[(bt * D_ + d) * 2], xi = xm[(bt * D_ + d) * 2 + 1];
    float wr = Wre[d * D_ + e], wi = Wim[d * D_ + e];
    ar += xr * wr - xi * wi;
    ai += xr * wi + xi * wr;
  }
  Xf[i * 2] = ar; Xf[i * 2 + 1] = ai;
}

// 8) flux scan over T with a_flux = exp(-softplus(lamf_re)+i lamf_im)
__global__ void k_scan(const float* __restrict__ Xf,
                       const float* __restrict__ lamf_re,
                       const float* __restrict__ lamf_im,
                       float* __restrict__ flux) {
  int i = threadIdx.x;
  if (i >= B_ * D_) return;
  int e = i & 63, b = i >> 6;
  float mag = expf(-softplusf(lamf_re[e]));
  float ar = mag * cosf(lamf_im[e]);
  float ai = mag * sinf(lamf_im[e]);
  float ur = 0.f, ui = 0.f;
  for (int t = 0; t < T_; ++t) {
    int idx = (b * T_ + t) * D_ + e;
    float nr = ar * ur - ai * ui + Xf[idx * 2];
    float ni = ar * ui + ai * ur + Xf[idx * 2 + 1];
    ur = nr; ui = ni;
    flux[idx * 2] = ur; flux[idx * 2 + 1] = ui;
  }
}

// 9) source, gate, decay, op_forcing per (b,t,d)
__global__ void k_btd(const float* __restrict__ flux,
                      const float* __restrict__ Wpre, const float* __restrict__ Wpim,
                      const float* __restrict__ Wg, const float* __restrict__ bg,
                      const float* __restrict__ lam_re, const float* __restrict__ lam_im,
                      const float* __restrict__ dt,
                      float* __restrict__ src, float* __restrict__ gate,
                      float* __restrict__ dec, float* __restrict__ opf) {
  int i = blockIdx.x * 256 + threadIdx.x;
  if (i >= BTD_) return;
  int d = i & 63, bt = i >> 6;
  int t = bt & 15;
  float sr = 0.f, si = 0.f, gsum = bg[d];
  for (int e = 0; e < D_; ++e) {
    float fr = flux[(bt * D_ + e) * 2], fi = flux[(bt * D_ + e) * 2 + 1];
    float wr = Wpre[e * D_ + d], wi = Wpim[e * D_ + d];
    sr += fr * wr - fi * wi;
    si += fr * wi + fi * wr;
    gsum += fr * Wg[e * D_ + d] + fi * Wg[(64 + e) * D_ + d];
  }
  src[i * 2] = sr; src[i * 2 + 1] = si;
  gate[i] = sigmoidf_(gsum);
  float lr = -softplusf(lam_re[d]);
  float li = lam_im[d];
  float dtv = dt[t];
  float er = expf(lr * dtv);
  float dr = er * cosf(li * dtv), di = er * sinf(li * dtv);
  dec[i * 2] = dr; dec[i * 2 + 1] = di;
  float nr = dr - 1.f, ni = di;
  float den = 1.f / (lr * lr + li * li);
  opf[i * 2] = (nr * lr + ni * li) * den;
  opf[i * 2 + 1] = (ni * lr - nr * li) * den;
}

// ---------------------------------------------------------------------------
// 10) Temporal pscan + forcing + noise (noise on output only)
// ---------------------------------------------------------------------------
__global__ __launch_bounds__(256) void k_pscan(const float* __restrict__ xeig,
                                               const float* __restrict__ gate,
                                               const float* __restrict__ src,
                                               const float* __restrict__ dec,
                                               const float* __restrict__ opf,
                                               const float* __restrict__ dt,
                                               float* __restrict__ u,
                                               half_t* __restrict__ u16) {
  int i = blockIdx.x * 256 + threadIdx.x;
  int d = i & 63;
  int hw = (i >> 6) & (HW_ - 1);
  int b = i >> 18;
  float ur = 0.f, ui = 0.f;
  for (int t = 0; t < T_; ++t) {
    size_t n = (size_t)((b * T_ + t) * HW_ + hw);
    int k = (b * T_ + t) * D_ + d;
    float xr = xeig[n * C2_ + d], xi = xeig[n * C2_ + 64 + d];
    float g = gate[k];
    float fr = xr * g + src[k * 2] * (1.f - g);
    float fi = xi * g + src[k * 2 + 1] * (1.f - g);
    float pr = opf[k * 2], pi = opf[k * 2 + 1];
    float tr = fr * pr - fi * pi, ti = fr * pi + fi * pr;
    float dr = dec[k * 2], di = dec[k * 2 + 1];
    float nr = dr * ur - di * ui + tr;
    float ni = dr * ui + di * ur + ti;
    ur = nr; ui = ni;
    unsigned s0 = pcg_hash((unsigned)(n * 64u + d) * 2u + 1u);
    unsigned s1 = pcg_hash((unsigned)(n * 64u + d) * 2u + 2u);
    float u1 = (float)(s0 >> 8) * 5.9604645e-8f + 1e-12f;
    float u2 = (float)(s1 >> 8) * 5.9604645e-8f;
    float rad = sqrtf(-2.f * logf(u1)) * 0.70710678f;
    float epr = rad * cosf(6.2831853f * u2);
    float epi = rad * sinf(6.2831853f * u2);
    float sc = 0.01f * sqrtf(dt[t]);
    float mr = 1.f + sc * epr, mi = sc * epi;
    float wr = ur * mr - ui * mi;
    float wi = ur * mi + ui * mr;
    u[n * C2_ + d] = wr;        u[n * C2_ + 64 + d] = wi;
    u16[n * C2_ + d] = (half_t)wr;  u16[n * C2_ + 64 + d] = (half_t)wi;
  }
}

// ---------------------------------------------------------------------------
// 11) LayerNorm over 128-wide rows; one wave32 per row
// ---------------------------------------------------------------------------
__global__ __launch_bounds__(256) void k_ln2(const float* __restrict__ y,
                                             const float* __restrict__ gamma,
                                             const float* __restrict__ beta,
                                             float* __restrict__ yn,
                                             half_t* __restrict__ feat) {
  int row = blockIdx.x * 8 + (threadIdx.x >> 5);
  int lane = threadIdx.x & 31;
  const float* yr = y + (size_t)row * C2_;
  f32x4 v = ((const f32x4*)yr)[lane];
  float s = v.x + v.y + v.z + v.w;
  float ss = v.x * v.x + v.y * v.y + v.z * v.z + v.w * v.w;
  for (int off = 16; off >= 1; off >>= 1) {
    s  += __shfl_xor(s, off);
    ss += __shfl_xor(ss, off);
  }
  float mean = s * (1.f / C2_);
  float var  = ss * (1.f / C2_) - mean * mean;
  float inv  = rsqrtf(var + 1e-5f);
  float* yno = yn + (size_t)row * C2_;
  half_t* fo = feat + (size_t)row * C2_;
  #pragma unroll
  for (int k = 0; k < 4; ++k) {
    int c = lane * 4 + k;
    float o = (v[k] - mean) * inv * gamma[c] + beta[c];
    yno[c] = o;
    fo[c] = (half_t)o;
  }
}

// ---------------------------------------------------------------------------
// 12) Fused MoE with DOUBLE-BUFFERED expert weights (305 KB of WGP LDS):
//     expert e+1's 128 KB DMA overlaps expert e's WMMA stream.
// ---------------------------------------------------------------------------
#define MOE_SMEM (32768 + 4*65536 + 8192 + 2048)   /* 305152 B */
__global__ __launch_bounds__(256) void k_moe(const half_t* __restrict__ feat,
                                             const float* __restrict__ yn,
                                             const float* __restrict__ Wr,
                                             const float* __restrict__ br,
                                             const float* __restrict__ b1,
                                             const float* __restrict__ b2,
                                             const half_t* __restrict__ W1T,
                                             const half_t* __restrict__ W2T,
                                             float* __restrict__ outc) {
  extern __shared__ char smem[];
  half_t* feat_s = (half_t*)smem;                  // 128x128
  half_t* w1b[2] = { feat_s + 16384, feat_s + 16384 + 32768 };
  half_t* w2b[2] = { feat_s + 16384 + 2*32768, feat_s + 16384 + 3*32768 };
  half_t* hst    = feat_s + 16384 + 4*32768;       // 8 waves x (16x32)
  float*  p_s    = (float*)(hst + 8 * 512);        // 128 x 4

  int tid = threadIdx.x, lane = tid & 31, wave = tid >> 5;
  size_t n0 = (size_t)blockIdx.x * 128;

  auto issue_expert = [&](int e, int setidx) {
    const half_t* w1g = W1T + (size_t)e * HID_ * C2_;
    const half_t* w2g = W2T + (size_t)e * C2_ * HID_;
    half_t* d1 = w1b[setidx];
    half_t* d2 = w2b[setidx];
    for (int i = tid; i < 4096; i += 256) async_copy_b128(d1 + i * 8, w1g + (size_t)i * 8);
    for (int i = tid; i < 4096; i += 256) async_copy_b128(d2 + i * 8, w2g + (size_t)i * 8);
  };

  // stage feature tile + expert-0 weights together
  for (int i = tid; i < 2048; i += 256)
    async_copy_b128(feat_s + i * 8, feat + n0 * C2_ + (size_t)i * 8);
  issue_expert(0, 0);
  async_wait0();
  __syncthreads();

  if (tid < 128) {
    float lg[NE_];
    for (int e = 0; e < NE_; ++e) lg[e] = br[e];
    for (int c = 0; c < C2_; ++c) {
      float f = (float)feat_s[tid * C2_ + c];
      for (int e = 0; e < NE_; ++e) lg[e] += f * Wr[c * NE_ + e];
    }
    float mx = fmaxf(fmaxf(lg[0], lg[1]), fmaxf(lg[2], lg[3]));
    float sm = 0.f;
    for (int e = 0; e < NE_; ++e) { lg[e] = expf(lg[e] - mx); sm += lg[e]; }
    float is = 1.f / sm;
    for (int e = 0; e < NE_; ++e) p_s[tid * NE_ + e] = lg[e] * is;
  }
  __syncthreads();

  const half_t* atile = feat_s + wave * 16 * C2_;
  half_t* hs = hst + wave * 512;
  v8f dacc[8] = {};

  for (int e = 0; e < NE_; ++e) {
    int cur = e & 1;
    if (e + 1 < NE_) issue_expert(e + 1, cur ^ 1);   // DMA overlaps compute
    half_t* w1_s = w1b[cur];
    half_t* w2_s = w2b[cur];

    v8f oacc[8] = {};
    for (int kc = 0; kc < 8; ++kc) {
      v8f hacc[2] = {};
      for (int ko = 0; ko < C2_; ko += 32) {
        v16h af = load_a_frag(atile, C2_, ko, lane);
        #pragma unroll
        for (int j = 0; j < 2; ++j) {
          v16h bf = load_b_frag(w1_s, C2_, kc * 32 + j * 16, ko, lane);
          hacc[j] = wmma16(af, bf, hacc[j]);
        }
      }
      #pragma unroll
      for (int j = 0; j < 2; ++j)
        #pragma unroll
        for (int r = 0; r < 8; ++r) {
          int row = r + (lane >> 4) * 8;
          int col = j * 16 + (lane & 15);
          float hv = hacc[j][r] + b1[e * HID_ + kc * 32 + col];
          hs[row * 32 + col] = (half_t)geluf_(hv);
        }
      __threadfence_block();
      v16h ha = load_a_frag(hs, 32, 0, lane);
      #pragma unroll
      for (int j2 = 0; j2 < 8; ++j2) {
        v16h bf2 = load_b_frag(w2_s, HID_, j2 * 16, kc * 32, lane);
        oacc[j2] = wmma16(ha, bf2, oacc[j2]);
      }
    }
    #pragma unroll
    for (int r = 0; r < 8; ++r) {
      int rowl = wave * 16 + r + (lane >> 4) * 8;
      float pe = p_s[rowl * NE_ + e];
      #pragma unroll
      for (int j = 0; j < 8; ++j) dacc[j][r] += pe * oacc[j][r];
    }
    async_wait0();     // next expert's weights landed
    __syncthreads();
  }

  #pragma unroll
  for (int j = 0; j < 8; ++j)
    #pragma unroll
    for (int r = 0; r < 8; ++r) {
      int rowl = wave * 16 + r + (lane >> 4) * 8;
      int col = j * 16 + (lane & 15);
      size_t n = n0 + rowl;
      float bterm = 0.f;
      for (int e = 0; e < NE_; ++e) bterm += p_s[rowl * NE_ + e] * b2[e * C2_ + col];
      outc[n * C2_ + col] = yn[n * C2_ + col] + dacc[j][r] + bterm;
    }
}

// ---------------------------------------------------------------------------
// 13) Final: out = x2 + outc, transposed to (2,B,T,D,H,W) via LDS tile
// ---------------------------------------------------------------------------
__global__ __launch_bounds__(256) void k_final(const float* __restrict__ outc,
                                               const float* __restrict__ x2,
                                               float* __restrict__ out) {
  __shared__ float lds[64 * 130];
  int hw0 = blockIdx.x * 64;
  int bt  = blockIdx.y;
  for (int it = 0; it < 32; ++it) {
    int idx = threadIdx.x + it * 256;
    int hwl = idx >> 7, c = idx & 127;
    lds[hwl * 130 + c] = outc[((size_t)(bt * HW_ + hw0 + hwl)) * C2_ + c];
  }
  __syncthreads();
  for (int it = 0; it < 32; ++it) {
    int idx = threadIdx.x + it * 256;
    int c = idx >> 6, hwl = idx & 63;
    float v = lds[hwl * 130 + c] + x2[(bt * C2_ + c) * HW_ + hw0 + hwl];
    int part = c >> 6, d = c & 63;
    out[(size_t)part * OUTPART_ + (size_t)(bt * D_ + d) * HW_ + hw0 + hwl] = v;
  }
}

// ---------------------------------------------------------------------------
// Launch
// ---------------------------------------------------------------------------
extern "C" void kernel_launch(void* const* d_in, const int* in_sizes, int n_in,
                              void* d_out, int out_size, void* d_ws, size_t ws_size,
                              hipStream_t stream) {
  (void)in_sizes; (void)n_in; (void)out_size; (void)ws_size;
  const float* x_re   = (const float*)d_in[0];
  const float* x_im   = (const float*)d_in[1];
  const float* dt     = (const float*)d_in[2];
  const float* gamma_s= (const float*)d_in[3];
  const float* beta_s = (const float*)d_in[4];
  const float* conv_w = (const float*)d_in[5];
  const float* conv_b = (const float*)d_in[6];
  const float* metric = (const float*)d_in[7];
  const float* gamma_t= (const float*)d_in[8];
  const float* beta_t = (const float*)d_in[9];
  const float* E_re   = (const float*)d_in[10];
  const float* E_im   = (const float*)d_in[11];
  const float* Dm_re  = (const float*)d_in[12];
  const float* Dm_im  = (const float*)d_in[13];
  const float* lamf_re= (const float*)d_in[14];
  const float* lamf_im= (const float*)d_in[15];
  const float* Win_re = (const float*)d_in[16];
  const float* Win_im = (const float*)d_in[17];
  const float* Wproj_re=(const float*)d_in[18];
  const float* Wproj_im=(const float*)d_in[19];
  const float* Wg     = (const float*)d_in[20];
  const float* bg     = (const float*)d_in[21];
  const float* lam_re = (const float*)d_in[22];
  const float* lam_im = (const float*)d_in[23];
  const float* Wr     = (const float*)d_in[24];
  const float* br     = (const float*)d_in[25];
  const float* W1     = (const float*)d_in[26];
  const float* b1     = (const float*)d_in[27];
  const float* W2     = (const float*)d_in[28];
  const float* b2     = (const float*)d_in[29];

  char* ws = (char*)d_ws;
  size_t off = 0;
  auto carve = [&](size_t bytes) { char* p = ws + off; off += (bytes + 255) & ~(size_t)255; return p; };
  half_t* xnT    = (half_t*)carve((size_t)NT_ * C2_ * 2);   // token-major LN1 out (reused as u16)
  float*  x2     = (float*) carve((size_t)NT_ * C2_ * 4);
  half_t* x2feat = (half_t*)carve((size_t)NT_ * C2_ * 2);   // reused as feat
  float*  xeig   = (float*) carve((size_t)NT_ * C2_ * 4);   // reused as y
  float*  u      = (float*) carve((size_t)NT_ * C2_ * 4);   // reused as outc
  float*  yn     = (float*) carve((size_t)NT_ * C2_ * 4);
  half_t* MET    = (half_t*)carve(C2_ * C2_ * 2);
  half_t* MDT    = (half_t*)carve(C2_ * C2_ * 2);
  half_t* WmT    = (half_t*)carve((size_t)9 * C2_ * C2_ * 2);
  half_t* W1T    = (half_t*)carve((size_t)NE_ * HID_ * C2_ * 2);
  half_t* W2T    = (half_t*)carve((size_t)NE_ * C2_ * HID_ * 2);
  float*  xm     = (float*) carve(BTD_ * 2 * 4);
  float*  Xf     = (float*) carve(BTD_ * 2 * 4);
  float*  flux   = (float*) carve(BTD_ * 2 * 4);
  float*  srcb   = (float*) carve(BTD_ * 2 * 4);
  float*  gate   = (float*) carve(BTD_ * 4);
  float*  dec    = (float*) carve(BTD_ * 2 * 4);
  float*  opf    = (float*) carve(BTD_ * 2 * 4);
  half_t* u16  = xnT;      // alias: conv input dead after k_conv
  float*  y    = xeig;     // alias: x_eigen dead after k_pscan
  half_t* feat = x2feat;   // alias: x2feat dead after first gemm
  float*  outc = u;        // alias: u (f32) dead after second gemm
  float*  out  = (float*)d_out;

  // weight prep
  k_build_cmat<<<64, 256, 0, stream>>>(E_re, E_im, MET);
  k_build_cmat<<<64, 256, 0, stream>>>(Dm_re, Dm_im, MDT);
  k_trans_convw<<<(9*C2_*C2_ + 255)/256, 256, 0, stream>>>(conv_w, WmT);
  k_trans_w<<<(NE_*C2_*HID_ + 255)/256, 256, 0, stream>>>(W1, W1T, C2_, HID_, NE_*C2_*HID_);
  k_trans_w<<<(NE_*HID_*C2_ + 255)/256, 256, 0, stream>>>(W2, W2T, HID_, C2_, NE_*HID_*C2_);

  // pipeline
  k_ln1<<<NT_/256, 256, 0, stream>>>(x_re, x_im, gamma_s, beta_s, xnT);
  k_conv<<<dim3(HW_/CPX_, B_*T_), 256, CONV_SMEM, stream>>>(xnT, WmT, conv_b, metric,
                                                            x_re, x_im, x2);
  k_pack<<<dim3(64, B_*T_), 256, 0, stream>>>(x2, x2feat);
  k_gemm128<<<NT_/128, 256, 0, stream>>>(x2feat, MET, xeig);
  k_mean<<<B_*T_, 256, 0, stream>>>(xeig, xm);
  k_xflux<<<(BTD_+255)/256, 256, 0, stream>>>(xm, Win_re, Win_im, Xf);
  k_scan<<<1, 128, 0, stream>>>(Xf, lamf_re, lamf_im, flux);
  k_btd<<<(BTD_+255)/256, 256, 0, stream>>>(flux, Wproj_re, Wproj_im, Wg, bg,
                                            lam_re, lam_im, dt, srcb, gate, dec, opf);
  k_pscan<<<(B_*HW_*D_)/256, 256, 0, stream>>>(xeig, gate, srcb, dec, opf, dt, u, u16);
  k_gemm128<<<NT_/128, 256, 0, stream>>>(u16, MDT, y);
  k_ln2<<<NT_/8, 256, 0, stream>>>(y, gamma_t, beta_t, yn, feat);
  k_moe<<<NT_/128, 256, MOE_SMEM, stream>>>(feat, yn, Wr, br, b1, b2, W1T, W2T, outc);
  k_final<<<dim3(64, B_*T_), 256, 0, stream>>>(outc, x2, out);
}
